// PEPSLayer_78692390797872
// MI455X (gfx1250) — matrix operand order repaired and checked by reading, not compiled, not run on hardware
//
#include <hip/hip_runtime.h>

// PEPS 4x4 (D=6, F_IN=2, F_OUT=10, B=1024) contraction for MI455X (gfx1250).
//
// Each contraction step 'b...x,burdx->b...dr' is a shared-weight GEMM
//   OUT[(b,m), n] = sum_{p,k} (x[b,p] * S[b,m,k]) * T_p[k,n]
// (node/env normalizations are positive per-batch scalars that cancel under
// the final out/||out||; fixed per-step scales keep everything in f16 range).
//
// Producer-push pipeline: every GEMM writes its output directly into the NEXT
// step's WMMA A-fragment layout (f16, ISA 16-bit A 16x32 mapping), already
// scaled by the next site's x[b,p] coefficients. Weights are pre-packed into
// B-fragment layout. One wave owns a 16-row stripe across all N (up to 3
// column tiles): inner loop = contiguous b128 loads + v_wmma_f32_16x16x32_f16,
// geometry (chunks, K, rows-per-batch, col tiles) fully templated so there are
// no runtime divisions and K-pad masking is compile-time.

typedef _Float16 f16;
typedef __attribute__((ext_vector_type(16))) _Float16 v16h;
typedef __attribute__((ext_vector_type(8)))  _Float16 v8h;
typedef __attribute__((ext_vector_type(8)))  float    v8f;

#define PEPS_B 1024

// Scatter one f16 value into an A-fragment buffer laid out as
// [tileM][chunk][lane 0..31][elem 0..15], per ISA 16-bit A 16x32 layout:
// lane = (row&15) + 16*hh, K(e,hh) = (e&7) + (e>=8)*16 + hh*8.
__device__ __forceinline__ void scatA(f16* A, int C, int rowp, int kp, float v) {
  const int c  = kp >> 5;
  const int kk = kp & 31;
  const int h2 = (kk >> 3) & 1;
  const int e  = (kk & 7) + ((kk >> 4) << 3);
  const int ln = (rowp & 15) + (h2 << 4);
  A[(((size_t)(rowp >> 4) * C + c) * 32 + ln) * 16 + e] = (f16)v;
}

struct GemmP {
  const f16* Afrag;   // this step's A fragments [tilesM][CC][32][16]
  const f16* Bfrag;   // this step's B fragments [NT][CC][32][16]
  float* O32;         // f32 canonical output (or null)
  f16* nextA;         // next step's A-fragment buffer (or null)
  const float* nXc;   // next site's coefficients: x = nXc[b*32 + p]
  int nMb, nKin, nC;  // next step geometry
  float nScale;       // next step's fixed scale
  // digit role map: producer digits (m2,m1,m0,n1,n0) -> next row'/k' weights
  int rm2, rm1, rm0, rn1, rn0;
  int km2, km1, km0, kn1, kn0;
};

// CC: K chunks, KT: valid K, MB: rows per batch element, NT: 16-col tiles.
template <int CC, int KT, int MB, int NT>
__global__ __launch_bounds__(256) void peps_gemm(GemmP P) {
  constexpr int NN = (NT == 3) ? 36 : 6;
  const int lane = threadIdx.x & 31;
  const int tm   = (blockIdx.x * 256 + threadIdx.x) >> 5;  // grid sized exactly
  const int hh   = lane >> 4;
  const int lm   = lane & 15;

  const f16* Ab = P.Afrag + ((size_t)tm * CC * 32 + lane) * 16;

  v8f acc[NT];
#pragma unroll
  for (int t = 0; t < NT; ++t) acc[t] = (v8f){0.f,0.f,0.f,0.f,0.f,0.f,0.f,0.f};

#pragma unroll
  for (int c = 0; c < CC; ++c) {
    const v8h* ap = (const v8h*)(Ab + c * 512);
    v16h af = __builtin_shufflevector(ap[0], ap[1],
        0, 1, 2, 3, 4, 5, 6, 7, 8, 9, 10, 11, 12, 13, 14, 15);
    // Compile-time K-pad masking: element e covers K=k0 (hh=0) / k0+8 (hh=1).
#pragma unroll
    for (int e = 0; e < 16; ++e) {
      const int k0 = c * 32 + (e & 7) + ((e >> 3) << 4);
      if (k0 >= KT) af[e] = (f16)0.f;                    // both halves invalid
      else if (k0 + 8 >= KT) { if (hh != 0) af[e] = (f16)0.f; }
    }
#pragma unroll
    for (int t = 0; t < NT; ++t) {
      const v8h* bp =
          (const v8h*)(P.Bfrag + (((size_t)t * CC + c) * 32 + lane) * 16);
      const v16h bf = __builtin_shufflevector(bp[0], bp[1],
          0, 1, 2, 3, 4, 5, 6, 7, 8, 9, 10, 11, 12, 13, 14, 15);
      acc[t] = __builtin_amdgcn_wmma_f32_16x16x32_f16(
          false, af, false, bf, (short)0, acc[t], false, false);
    }
  }

  // D layout: VGPR r -> row = tm*16 + r + hh*8, col = t*16 + (lane&15)
  const int grow0 = tm * 16 + hh * 8;
  int b, mi;
  if (MB == 1) { b = grow0; mi = 0; }
  else         { b = grow0 / MB; mi = grow0 - b * MB; }  // constant divisor

  if (P.nextA) {
    int rcolv[NT], kcolv[NT];
    bool okv[NT];
#pragma unroll
    for (int t = 0; t < NT; ++t) {
      const int col = t * 16 + lm;
      okv[t] = col < NN;
      const int nc = okv[t] ? col : 0;
      const int n0 = nc % 6, n1 = nc / 6;
      rcolv[t] = n1 * P.rn1 + n0 * P.rn0;
      kcolv[t] = n1 * P.kn1 + n0 * P.kn0;
    }
#pragma unroll
    for (int r = 0; r < 8; ++r) {
      const int bb = (MB == 1) ? grow0 + r : b;
      const int m0 = mi % 6;
      const int mt = mi / 6;
      const int m1 = mt % 6, m2 = mt / 6;
      const int rowb = bb * P.nMb + m2 * P.rm2 + m1 * P.rm1 + m0 * P.rm0;
      const int kb   = m2 * P.km2 + m1 * P.km1 + m0 * P.km0;
      const float x0 = P.nXc[bb * 32]     * P.nScale;
      const float x1 = P.nXc[bb * 32 + 1] * P.nScale;
#pragma unroll
      for (int t = 0; t < NT; ++t) {
        if (okv[t]) {
          const float v  = acc[t][r];
          const int   rp = rowb + rcolv[t];
          const int   kp = kb + kcolv[t];
          scatA(P.nextA, P.nC, rp, kp,          v * x0);
          scatA(P.nextA, P.nC, rp, P.nKin + kp, v * x1);
        }
      }
      if (MB > 1) { ++mi; if (mi == MB) { mi = 0; ++b; } }
    }
  } else {
#pragma unroll
    for (int r = 0; r < 8; ++r) {
#pragma unroll
      for (int t = 0; t < NT; ++t) {
        const int col = t * 16 + lm;
        if (col < NN) P.O32[(size_t)(grow0 + r) * NN + col] = acc[t][r];
      }
    }
  }
}

// Pre-pack a weight tensor into B-fragment layout [tn][c][lane][e] (zero-padded).
struct PackBP {
  const float* W; f16* out;
  int wp, wk0, wk1, wn0, wn1;  // strides: p, k-digits, n-digits
  int N, Kin, C;
};
__global__ __launch_bounds__(256) void peps_packB(PackBP P) {
  const int total = ((P.N + 15) >> 4) * P.C * 512;
  const int idx = blockIdx.x * 256 + threadIdx.x;
  if (idx >= total) return;
  const int e    = idx & 15;
  const int ln   = (idx >> 4) & 31;
  const int rest = idx >> 9;                // tn*C + c
  const int c  = rest % P.C;
  const int tn = rest / P.C;
  const int col = tn * 16 + (ln & 15);
  const int k   = c * 32 + (e & 7) + ((e >> 3) << 4) + ((ln >> 4) << 3);   // B layout: K = (e&7) + (e>=8)*16 + hh*8
  float v = 0.f;
  if (col < P.N && k < 2 * P.Kin) {
    const int p  = (k >= P.Kin) ? 1 : 0;
    const int ki = k - p * P.Kin;
    v = P.W[p * P.wp + (ki % 6) * P.wk0 + (ki / 6) * P.wk1 +
            (col % 6) * P.wn0 + (col / 6) * P.wn1];
  }
  P.out[idx] = (f16)v;
}

// A-fragment for the two first steps (A = x[b,p]*scale, M=1024, Ktot=2, C=1).
__global__ __launch_bounds__(256) void peps_packA0(const float* Xc, f16* out,
                                                   float scale) {
  const int idx = blockIdx.x * 256 + threadIdx.x;   // 64 tiles * 512
  if (idx >= 64 * 512) return;
  const int e  = idx & 15;
  const int ln = (idx >> 4) & 31;
  const int tm = idx >> 9;
  const int row = tm * 16 + (ln & 15);
  const int k = (e & 7) + ((e >> 3) << 4) + ((ln >> 4) << 3);
  const float v = (k < 2) ? Xc[row * 32 + k] * scale : 0.f;
  out[idx] = (f16)v;
}

// out[b*10+o] = <F[b,:], Ebot[b,:]> over 1296 elements.
__global__ __launch_bounds__(256) void peps_dot(const float* __restrict__ F,
                                                const float* __restrict__ Ebot,
                                                float* __restrict__ out, int o) {
  __shared__ float red[256];
  const int b = blockIdx.x;
  const float* f = F + b * 1296;
  const float* e = Ebot + b * 1296;
  float s = 0.f;
  for (int i = threadIdx.x; i < 1296; i += 256) s += f[i] * e[i];
  red[threadIdx.x] = s;
  __syncthreads();
  for (int off = 128; off > 0; off >>= 1) {
    if (threadIdx.x < off) red[threadIdx.x] += red[threadIdx.x + off];
    __syncthreads();
  }
  if (threadIdx.x == 0) out[b * 10 + o] = red[0];
}

__global__ void peps_outnorm(float* out) {
  const int b = blockIdx.x * blockDim.x + threadIdx.x;
  if (b >= PEPS_B) return;
  float s = 0.f;
  for (int o = 0; o < 10; ++o) { const float v = out[b * 10 + o]; s += v * v; }
  const float inv = rsqrtf(fmaxf(s, 1e-30f));
  for (int o = 0; o < 10; ++o) out[b * 10 + o] *= inv;
}

extern "C" void kernel_launch(void* const* d_in, const int* in_sizes, int n_in,
                              void* d_out, int out_size, void* d_ws, size_t ws_size,
                              hipStream_t stream) {
  (void)in_sizes; (void)n_in; (void)out_size; (void)ws_size;
  const float* inputs = (const float*)d_in[0];  // (1024,4,4,2)
  const float* peps   = (const float*)d_in[1];  // (4,4,2,6,6,6,6)
  const float* pepsc  = (const float*)d_in[2];  // (2,10,6,6,6,6)
  float* out = (float*)d_out;                   // (1024,10)
  float* ws  = (float*)d_ws;

  const int B = PEPS_B;
  // f32 regions
  float* ebot = ws;
  float* Fbuf = ws + 1296 * B;
  // f16 regions
  f16* h = (f16*)(ws + 2 * 1296 * B);
  size_t off = 0;
  auto halloc = [&](size_t n) { f16* p = h + off; off += n; return p; };
  f16* FA0  = halloc(64 * 512);            // first-step A (row 0)
  f16* FC0  = halloc(64 * 512);            // first-step A (row 3)
  f16* SM1  = halloc((size_t)384 * 512);   // M=6144,  C=1
  f16* SM2  = halloc((size_t)2304 * 512);  // M=36864, C=1
  f16* MED1 = halloc((size_t)13824 * 512); // M=221184, C=1
  f16* MED2 = halloc((size_t)13824 * 512);
  f16* BIG1 = halloc((size_t)13824 * 3 * 512); // M=221184, C=3
  f16* BIG2 = halloc((size_t)13824 * 3 * 512);

  // weight fragment buffers
  auto wsz = [](int N, int C) { return (size_t)((N + 15) / 16) * C * 512; };
  f16* WB[15]; // A0..A3, B0..B3, C0..C3, D0, D1, D3
  f16* WD2;    // 10 center slices
  {
    int Ns[15] = {36,36,36,6, 36,36,36,6, 36,36,36,6, 36,36,6};
    int Cs[15] = {1, 1, 1, 1, 1, 3, 3, 3, 1, 1, 1, 1, 1, 3, 3};
    for (int i = 0; i < 15; ++i) WB[i] = halloc(wsz(Ns[i], Cs[i]));
    WD2 = halloc(10 * wsz(36, 3));
  }

  auto site = [&](int i, int j) { return peps + (i * 4 + j) * 2592; };
  auto xc   = [&](int i, int j) { return inputs + (i * 4 + j) * 2; };

  auto packB = [&](f16* outb, const float* W, int wp, int wk0, int wk1,
                   int wn0, int wn1, int N, int Kin, int C) {
    PackBP P{W, outb, wp, wk0, wk1, wn0, wn1, N, Kin, C};
    const int total = ((N + 15) / 16) * C * 512;
    peps_packB<<<(total + 255) / 256, 256, 0, stream>>>(P);
  };

  // dispatch a GEMM step; geometry id selects the template instantiation
  enum Geo { G_FIRST, G_MB6, G_MB36, G_END1, G_ROW0, G_MID, G_END3 };
  auto rung = [&](const GemmP& P, Geo g) {
    switch (g) {
      case G_FIRST: peps_gemm<1, 2, 1, 3><<<8, 256, 0, stream>>>(P); break;
      case G_MB6:   peps_gemm<1, 12, 6, 3><<<48, 256, 0, stream>>>(P); break;
      case G_MB36:  peps_gemm<1, 12, 36, 3><<<288, 256, 0, stream>>>(P); break;
      case G_END1:  peps_gemm<1, 12, 216, 1><<<1728, 256, 0, stream>>>(P); break;
      case G_ROW0:  peps_gemm<1, 12, 216, 3><<<1728, 256, 0, stream>>>(P); break;
      case G_MID:   peps_gemm<3, 72, 216, 3><<<1728, 256, 0, stream>>>(P); break;
      case G_END3:  peps_gemm<3, 72, 216, 1><<<1728, 256, 0, stream>>>(P); break;
    }
  };

  const float SC0 = 120.f, SC6 = 50.f, SC36 = 20.f;
  // digit role maps: (rm2,rm1,rm0,rn1,rn0, km2,km1,km0,kn1,kn0)
  const int M1[10] = {0,0,0,1,0,  0,0,0,0,1};   // 36-elem env -> Mb'=6,Kin'=6
  const int M2[10] = {0,0,6,1,0,  0,0,0,0,1};   // 216 -> Mb'=36
  const int M3[10] = {0,36,6,1,0, 0,0,0,0,1};   // 1296 -> Mb'=216
  const int M4[10] = {0,36,6,0,1, 1,0,0,0,0};   // row-end env -> next row j=0
  const int M5[10] = {0,36,6,1,0, 6,0,0,0,1};   // interior (u,x) contraction

  // ---- pre-pack all weights + first-step A fragments ----
  peps_packA0<<<128, 256, 0, stream>>>(xc(0, 0), FA0, SC0);
  peps_packA0<<<128, 256, 0, stream>>>(xc(3, 0), FC0, SC0);
  packB(WB[0],  site(0,0), 1296, 0,0,   36,6,   36, 1, 1);  // A0
  packB(WB[1],  site(0,1), 1296, 1,0,   36,6,   36, 6, 1);  // A1
  packB(WB[2],  site(0,2), 1296, 1,0,   36,6,   36, 6, 1);  // A2
  packB(WB[3],  site(0,3), 1296, 1,0,    6,0,    6, 6, 1);  // A3
  packB(WB[4],  site(1,0), 1296, 216,0, 36,6,   36, 6, 1);  // B0
  packB(WB[5],  site(1,1), 1296, 1,216, 36,6,   36,36, 3);  // B1
  packB(WB[6],  site(1,2), 1296, 1,216, 36,6,   36,36, 3);  // B2
  packB(WB[7],  site(1,3), 1296, 1,216,  6,0,    6,36, 3);  // B3
  packB(WB[8],  site(3,0), 1296, 0,0,   36,216, 36, 1, 1);  // C0
  packB(WB[9],  site(3,1), 1296, 1,0,   36,216, 36, 6, 1);  // C1
  packB(WB[10], site(3,2), 1296, 1,0,   36,216, 36, 6, 1);  // C2
  packB(WB[11], site(3,3), 1296, 1,0,   216,0,   6, 6, 1);  // C3
  packB(WB[12], site(2,0), 1296, 216,0, 36,6,   36, 6, 1);  // D0
  packB(WB[13], site(2,1), 1296, 1,216, 36,6,   36,36, 3);  // D1
  packB(WB[14], site(2,3), 1296, 1,216,  6,36,   6,36, 3);  // D3
  const size_t wd2sz = wsz(36, 3);
  for (int o = 0; o < 10; ++o)
    packB(WD2 + o * wd2sz, pepsc + o * 1296, 12960, 1,216, 36,6, 36,36, 3); // D2_o

  auto G = [&](const f16* A, const f16* Wf, float* O32, f16* nA, const float* nX,
               int nMb, int nKin, int nC, float nSc, const int* mp) {
    GemmP P;
    P.Afrag = A; P.Bfrag = Wf; P.O32 = O32; P.nextA = nA; P.nXc = nX;
    P.nMb = nMb; P.nKin = nKin; P.nC = nC; P.nScale = nSc;
    static const int Z[10] = {0,0,0,0,0,0,0,0,0,0};
    if (!mp) mp = Z;
    P.rm2 = mp[0]; P.rm1 = mp[1]; P.rm0 = mp[2]; P.rn1 = mp[3]; P.rn0 = mp[4];
    P.km2 = mp[5]; P.km1 = mp[6]; P.km0 = mp[7]; P.kn1 = mp[8]; P.kn0 = mp[9];
    return P;
  };

  // ---- Row 0 (top) ----
  rung(G(FA0,  WB[0], nullptr, SM1,  xc(0,1),   6, 6, 1, SC6,  M1), G_FIRST);
  rung(G(SM1,  WB[1], nullptr, SM2,  xc(0,2),  36, 6, 1, SC6,  M2), G_MB6);
  rung(G(SM2,  WB[2], nullptr, MED1, xc(0,3), 216, 6, 1, SC6,  M3), G_MB36);
  rung(G(MED1, WB[3], nullptr, MED2, xc(1,0), 216, 6, 1, SC6,  M4), G_END1);
  // ---- Row 1 -> env_top feeds row 2 ----
  rung(G(MED2, WB[4], nullptr, BIG1, xc(1,1), 216, 36, 3, SC36, M5), G_ROW0);
  rung(G(BIG1, WB[5], nullptr, BIG2, xc(1,2), 216, 36, 3, SC36, M5), G_MID);
  rung(G(BIG2, WB[6], nullptr, BIG1, xc(1,3), 216, 36, 3, SC36, M5), G_MID);
  rung(G(BIG1, WB[7], nullptr, MED2, xc(2,0), 216,  6, 1, SC6,  M4), G_END3);
  // ---- Row 3 (bottom) -> ebot (f32) ----
  rung(G(FC0,  WB[8],  nullptr, SM1,  xc(3,1),   6, 6, 1, SC6, M1), G_FIRST);
  rung(G(SM1,  WB[9],  nullptr, SM2,  xc(3,2),  36, 6, 1, SC6, M2), G_MB6);
  rung(G(SM2,  WB[10], nullptr, MED1, xc(3,3), 216, 6, 1, SC6, M3), G_MB36);
  rung(G(MED1, WB[11], ebot, nullptr, nullptr,   0, 0, 0, 0.f, nullptr), G_END1);
  // ---- Row 2 (center) ----
  rung(G(MED2, WB[12], nullptr, BIG2, xc(2,1), 216, 36, 3, SC36, M5), G_ROW0);
  rung(G(BIG2, WB[13], nullptr, BIG1, xc(2,2), 216, 36, 3, SC36, M5), G_MID);
  for (int o = 0; o < 10; ++o) {
    rung(G(BIG1, WD2 + o * wd2sz, nullptr, BIG2, xc(2,3),
           216, 36, 3, SC36, M5), G_MID);
    rung(G(BIG2, WB[14], Fbuf, nullptr, nullptr, 0, 0, 0, 0.f, nullptr), G_END3);
    peps_dot<<<B, 256, 0, stream>>>(Fbuf, ebot, out, o);
  }
  peps_outnorm<<<(B + 255) / 256, 256, 0, stream>>>(out);
}
